// SimpleMetaNet_89970974917154
// MI455X (gfx1250) — compile-verified, matrix operand order checked by reading, not run on hardware
//
#include <hip/hip_runtime.h>

typedef __attribute__((ext_vector_type(16))) _Float16 v16h;
typedef __attribute__((ext_vector_type(8)))  float    v8f;

#define TPB 256
#define WPB (TPB / 32)   // waves per block (wave32 on gfx1250)

// ---------------------------------------------------------------------------
// K1: per-element MLP via WMMA (computed transposed: h2^T = W2^T x h1^T so the
// result lands element-per-lane); writes logits into `logits` (== d_out) and
// per-block partials: max(logit), sum(grad^2).
// ---------------------------------------------------------------------------
__global__ __launch_bounds__(TPB)
void meta_logits_kernel(const float* __restrict__ grad,
                        const float* __restrict__ W1, const float* __restrict__ b1,
                        const float* __restrict__ W2, const float* __restrict__ b2,
                        const float* __restrict__ W3, const float* __restrict__ b3,
                        float* __restrict__ logits,
                        float* __restrict__ maxPart, float* __restrict__ ssqPart,
                        int N)
{
    const int tid  = threadIdx.x;
    const int lane = tid & 31;
    const int m    = lane & 15;            // this lane's element-in-tile / matrix row
    const int kh   = (lane & 16) ? 8 : 0;  // K-half per ISA 16-bit 16x32 operand layout

    // Per-lane element->K mapping (identical for A and B operands of this shape):
    //   k(e) = kh + (e<8 ? e : e+8)
    // Preload:
    //   w1h/b1h : W1,b1 at this lane's K slots, packed f16 (layer-1 pk_fma)
    //   Wt0/Wt1 : A operand = W2^T rows 0..15 / 16..31  (Wt[e] = W2[j(e)*32 + row])
    //   b2v/w3v : b2,W3 at this lane's K slots (layer-3 dot product, f32)
    v16h w1h, b1h, Wt0, Wt1;
    float b2v[16], w3v[16];
#pragma unroll
    for (int e = 0; e < 16; ++e) {
        int k  = kh + ((e < 8) ? e : e + 8);
        w1h[e] = (_Float16)W1[k];
        b1h[e] = (_Float16)b1[k];
        Wt0[e] = (_Float16)W2[k * 32 + m];        // W2^T rows 0..15, K-dim j=k(e)
        Wt1[e] = (_Float16)W2[k * 32 + 16 + m];   // W2^T rows 16..31
        b2v[e] = b2[k];
        w3v[e] = W3[k];
    }
    const float b3v = b3[0];

    float lmax = -3.402823466e+38f;
    float lssq = 0.0f;

    const int nWaves = gridDim.x * WPB;
    const int wave   = blockIdx.x * WPB + (tid >> 5);
    const int step   = nWaves * 32;                 // 2 tiles per wave iteration

    for (int base = wave * 32; base < N; base += step) {
        if (base + step < N)                        // wave-uniform
            __builtin_prefetch(&grad[base + step], 0, 3);

        const int iA = base + m;
        const int iB = base + 16 + m;
        float g0 = (iA < N) ? grad[iA] : 0.0f;
        float g1 = (iB < N) ? grad[iB] : 0.0f;
        if (lane < 16) {                            // count each element once
            if (iA < N) lssq += g0 * g0;
            if (iB < N) lssq += g1 * g1;
        }

        // Layer 1 (packed f16): h1 = relu(g*W1 + b1) -> B operand (h1^T layout).
        _Float16 gh0 = (_Float16)g0, gh1 = (_Float16)g1;
        v16h Ga, Gb;
#pragma unroll
        for (int e = 0; e < 16; ++e) { Ga[e] = gh0; Gb[e] = gh1; }
        v16h Ha = Ga * w1h + b1h;
        v16h Hb = Gb * w1h + b1h;
        const _Float16 z = (_Float16)0.0f;
#pragma unroll
        for (int e = 0; e < 16; ++e) {
            Ha[e] = (Ha[e] > z) ? Ha[e] : z;
            Hb[e] = (Hb[e] > z) ? Hb[e] : z;
        }

        // Layer 2: h2^T = W2^T x h1^T, four v_wmma_f32_16x16x32_f16.
        // D layout: lane L holds its own element m, rows k=kh+r (t0) / 16+kh+r (t1).
        v8f a0 = {}, a1 = {}, c0 = {}, c1 = {};
        a0 = __builtin_amdgcn_wmma_f32_16x16x32_f16(false, Wt0, false, Ha,
                                                    (short)0, a0, false, false);
        a1 = __builtin_amdgcn_wmma_f32_16x16x32_f16(false, Wt1, false, Ha,
                                                    (short)0, a1, false, false);
        c0 = __builtin_amdgcn_wmma_f32_16x16x32_f16(false, Wt0, false, Hb,
                                                    (short)0, c0, false, false);
        c1 = __builtin_amdgcn_wmma_f32_16x16x32_f16(false, Wt1, false, Hb,
                                                    (short)0, c1, false, false);

        // Layer 3: per-lane half-dot, then combine the two K-halves (1 shuffle).
        float pA = 0.0f, pB = 0.0f;
#pragma unroll
        for (int e = 0; e < 16; ++e) {
            float ha = (e < 8) ? a0[e] : a1[e - 8];
            float hb = (e < 8) ? c0[e] : c1[e - 8];
            pA += fmaxf(ha + b2v[e], 0.0f) * w3v[e];
            pB += fmaxf(hb + b2v[e], 0.0f) * w3v[e];
        }
        pA += __shfl_xor(pA, 16, 32);
        pB += __shfl_xor(pB, 16, 32);
        const float lgA = pA + b3v;
        const float lgB = pB + b3v;

        if (lane < 16) {                            // coalesced 16-float stores
            if (iA < N) { logits[iA] = lgA; lmax = fmaxf(lmax, lgA); }
            if (iB < N) { logits[iB] = lgB; lmax = fmaxf(lmax, lgB); }
        }
    }

    // Block reduction of (max, sumsq).
    __shared__ float smax[TPB];
    __shared__ float ssum[TPB];
    smax[tid] = lmax;
    ssum[tid] = lssq;
    __syncthreads();
    for (int s = TPB / 2; s > 0; s >>= 1) {
        if (tid < s) {
            smax[tid] = fmaxf(smax[tid], smax[tid + s]);
            ssum[tid] += ssum[tid + s];
        }
        __syncthreads();
    }
    if (tid == 0) {
        maxPart[blockIdx.x] = smax[0];
        ssqPart[blockIdx.x] = ssum[0];
    }
}

// ---------------------------------------------------------------------------
// K2: reduce per-block partials -> scal[0] = max logit, scal[1] = ||grad||.
// ---------------------------------------------------------------------------
__global__ __launch_bounds__(TPB)
void reduce_pass1(const float* __restrict__ maxPart, const float* __restrict__ ssqPart,
                  int P, float* __restrict__ scal)
{
    const int tid = threadIdx.x;
    float lmax = -3.402823466e+38f, lsum = 0.0f;
    for (int i = tid; i < P; i += TPB) {
        lmax = fmaxf(lmax, maxPart[i]);
        lsum += ssqPart[i];
    }
    __shared__ float smax[TPB];
    __shared__ float ssum[TPB];
    smax[tid] = lmax; ssum[tid] = lsum;
    __syncthreads();
    for (int s = TPB / 2; s > 0; s >>= 1) {
        if (tid < s) {
            smax[tid] = fmaxf(smax[tid], smax[tid + s]);
            ssum[tid] += ssum[tid + s];
        }
        __syncthreads();
    }
    if (tid == 0) {
        scal[0] = smax[0];
        scal[1] = sqrtf(ssum[0]);   // grad_norm
    }
}

// ---------------------------------------------------------------------------
// K3: e = exp(l - max); partial sums of e and (e*g)^2. float4, L2-resident.
// ---------------------------------------------------------------------------
__global__ __launch_bounds__(TPB)
void exp_sums_kernel(const float* __restrict__ logits, const float* __restrict__ grad,
                     const float* __restrict__ scal, int N,
                     float* __restrict__ sePart, float* __restrict__ s2Part)
{
    const int tid  = threadIdx.x;
    const int gid  = blockIdx.x * TPB + tid;
    const int strd = gridDim.x * TPB;
    const float maxl = scal[0];
    float se = 0.0f, s2 = 0.0f;

    const float4* l4 = (const float4*)logits;
    const float4* g4 = (const float4*)grad;
    const int nv = N >> 2;
    for (int i = gid; i < nv; i += strd) {
        float4 l = l4[i];
        float4 g = g4[i];
        float e0 = __expf(l.x - maxl), e1 = __expf(l.y - maxl);
        float e2 = __expf(l.z - maxl), e3 = __expf(l.w - maxl);
        se += (e0 + e1) + (e2 + e3);
        float m0 = e0 * g.x, m1 = e1 * g.y, m2 = e2 * g.z, m3 = e3 * g.w;
        s2 += (m0 * m0 + m1 * m1) + (m2 * m2 + m3 * m3);
    }
    for (int i = (nv << 2) + gid; i < N; i += strd) {   // tail
        float e = __expf(logits[i] - maxl);
        float eg = e * grad[i];
        se += e; s2 += eg * eg;
    }

    __shared__ float sse[TPB];
    __shared__ float ss2[TPB];
    sse[tid] = se; ss2[tid] = s2;
    __syncthreads();
    for (int s = TPB / 2; s > 0; s >>= 1) {
        if (tid < s) { sse[tid] += sse[tid + s]; ss2[tid] += ss2[tid + s]; }
        __syncthreads();
    }
    if (tid == 0) {
        sePart[blockIdx.x] = sse[0];
        s2Part[blockIdx.x] = ss2[0];
    }
}

// ---------------------------------------------------------------------------
// K4: final scalar C = rescale * dyn / sumexp,
// masked_norm = sqrt(S2)/sumexp, dyn = ||grad||/(masked_norm+eps) (or 1).
// ---------------------------------------------------------------------------
__global__ __launch_bounds__(TPB)
void reduce_pass2(const float* __restrict__ sePart, const float* __restrict__ s2Part,
                  int Q, const float* __restrict__ rescale, float* __restrict__ scal)
{
    const int tid = threadIdx.x;
    float se = 0.0f, s2 = 0.0f;
    for (int i = tid; i < Q; i += TPB) { se += sePart[i]; s2 += s2Part[i]; }
    __shared__ float sse[TPB];
    __shared__ float ss2[TPB];
    sse[tid] = se; ss2[tid] = s2;
    __syncthreads();
    for (int s = TPB / 2; s > 0; s >>= 1) {
        if (tid < s) { sse[tid] += sse[tid + s]; ss2[tid] += ss2[tid + s]; }
        __syncthreads();
    }
    if (tid == 0) {
        float sumexp      = sse[0];
        float masked_norm = sqrtf(ss2[0]) / sumexp;
        float eps         = 1e-8f;
        float dyn = (masked_norm > eps) ? (scal[1] / (masked_norm + eps)) : 1.0f;
        scal[2] = rescale[0] * dyn / sumexp;
    }
}

// ---------------------------------------------------------------------------
// K5: out[i] = C * exp(logit[i]-max) * grad[i]; logits read from d_out in place.
// ---------------------------------------------------------------------------
__global__ __launch_bounds__(TPB)
void finalize_kernel(float* __restrict__ out, const float* __restrict__ grad,
                     const float* __restrict__ scal, int N)
{
    const int gid  = blockIdx.x * TPB + threadIdx.x;
    const int strd = gridDim.x * TPB;
    const float maxl = scal[0];
    const float C    = scal[2];

    float4* o4 = (float4*)out;
    const float4* g4 = (const float4*)grad;
    const int nv = N >> 2;
    for (int i = gid; i < nv; i += strd) {
        float4 l = o4[i];
        float4 g = g4[i];
        float4 r;
        r.x = C * __expf(l.x - maxl) * g.x;
        r.y = C * __expf(l.y - maxl) * g.y;
        r.z = C * __expf(l.z - maxl) * g.z;
        r.w = C * __expf(l.w - maxl) * g.w;
        o4[i] = r;
    }
    for (int i = (nv << 2) + gid; i < N; i += strd) {   // tail
        float l = out[i];
        out[i] = C * __expf(l - maxl) * grad[i];
    }
}

// ---------------------------------------------------------------------------
extern "C" void kernel_launch(void* const* d_in, const int* in_sizes, int n_in,
                              void* d_out, int out_size, void* d_ws, size_t ws_size,
                              hipStream_t stream)
{
    const float* grad    = (const float*)d_in[0];
    const float* W1      = (const float*)d_in[1];
    const float* b1      = (const float*)d_in[2];
    const float* W2      = (const float*)d_in[3];
    const float* b2      = (const float*)d_in[4];
    const float* W3      = (const float*)d_in[5];
    const float* b3      = (const float*)d_in[6];
    const float* rescale = (const float*)d_in[7];
    float* out = (float*)d_out;
    const int N = in_sizes[0];

    const int eltsPerBlk = WPB * 32;                 // 256 elements / block / iter
    int P = (N + eltsPerBlk - 1) / eltsPerBlk;
    if (P > 2048) P = 2048;
    if (P < 1) P = 1;
    int Q = (N / 4 + TPB - 1) / TPB;
    if (Q > 2048) Q = 2048;
    if (Q < 1) Q = 1;

    // Tiny workspace layout (floats): maxPart[P] ssqPart[P] sePart[Q] s2Part[Q] scal[4]
    float* wsf  = (float*)d_ws;
    float* maxP = wsf;
    float* ssqP = wsf + P;
    float* seP  = wsf + 2 * P;
    float* s2P  = wsf + 2 * P + Q;
    float* scal = wsf + 2 * P + 2 * Q;

    meta_logits_kernel<<<P, TPB, 0, stream>>>(grad, W1, b1, W2, b2, W3, b3,
                                              out, maxP, ssqP, N);
    reduce_pass1<<<1, TPB, 0, stream>>>(maxP, ssqP, P, scal);
    exp_sums_kernel<<<Q, TPB, 0, stream>>>(out, grad, scal, N, seP, s2P);
    reduce_pass2<<<1, TPB, 0, stream>>>(seP, s2P, Q, rescale, scal);
    finalize_kernel<<<Q, TPB, 0, stream>>>(out, grad, scal, N);
}
